// ProbSparseAttention_38259568673263
// MI455X (gfx1250) — matrix operand order repaired and checked by reading, not compiled, not run on hardware
//
#include <hip/hip_runtime.h>

// ---------------------------------------------------------------------------
// ProbSparse attention for MI455X (gfx1250, wave32, WMMA 16x16x32 bf16,
// async global->LDS staging with ASYNCcnt).
//   1) gemm_xwt<1> x3 : Q/K/V projections (fp32 in, bf16 out in [B,H,L,dk])
//   2) attn_topk      : scores in 256KB LDS, top-1024 threshold via bisection,
//                       softmax-in-place, P@V with WMMA + async V staging
//   3) gemm_xwt<0>    : output projection (fp32 out)
// ---------------------------------------------------------------------------

typedef __attribute__((ext_vector_type(16))) __bf16 v16bf;
typedef __attribute__((ext_vector_type(8)))  float  v8f;

#define L_SEQ   2048
#define D_MODEL 1024
#define NH      16
#define D_HEAD  64
#define TOPK    1024

__device__ inline v8f wmma_bf16(v16bf a, v16bf b, v8f c) {
  return __builtin_amdgcn_wmma_f32_16x16x32_bf16(false, a, false, b,
                                                 (short)0, c, false, false);
}

// Generic pointer to __shared__ = {SHARED_BASE_hi, lds_offset}; low 32 bits
// are exactly the LDS byte offset the async-to-LDS VDST operand expects.
__device__ inline unsigned lds_off(const void* p) {
  return (unsigned)(unsigned long long)p;
}

// 16B/lane async copy global -> LDS (tracked by ASYNCcnt, not LOADcnt)
__device__ inline void async_cp_b128(unsigned lds_byte_off, const void* gaddr) {
  asm volatile("global_load_async_to_lds_b128 %0, %1, off"
               :: "v"(lds_byte_off), "v"(gaddr) : "memory");
}
__device__ inline void wait_async_le1() {
  asm volatile("s_wait_asynccnt 0x1" ::: "memory");
}
__device__ inline void wait_async_le0() {
  asm volatile("s_wait_asynccnt 0x0" ::: "memory");
}

// B fragment (32x16 bf16): lane half h holds contiguous K run [16h,16h+16)
// for column (lane&15); source row-major with K contiguous (i.e. W row n).
__device__ inline v16bf b_frag_f32(const float* __restrict__ W, int ld,
                                   int row, int k0, int lane) {
  const float* p = W + (size_t)row * ld + k0 + ((lane >> 4) << 4);
  v16bf b;
#pragma unroll
  for (int e = 0; e < 16; ++e) b[e] = (__bf16)p[e];
  return b;
}

__device__ inline v16bf a_frag_bf16(const __bf16* __restrict__ X, int ld,
                                    int row, int k0, int lane) {
  const __bf16* p0 = X + (size_t)row * ld + k0 + ((lane >> 4) << 4);
  v16bf a;
#pragma unroll
  for (int j = 0; j < 8; ++j) { a[j] = p0[j]; a[j + 8] = p0[8 + j]; }
  return a;
}

__device__ inline v16bf b_frag_bf16(const __bf16* __restrict__ W, int ld,
                                    int row, int k0, int lane) {
  const __bf16* p = W + (size_t)row * ld + k0 + ((lane >> 4) << 4);
  v16bf b;
#pragma unroll
  for (int e = 0; e < 16; ++e) b[e] = p[e];
  return b;
}

// Y[M,N] = X[M,K] @ W[N,K]^T + bias.  One wave = 32x32 tile (4 WMMA / k-step).
// All 8 waves of a block share m0, so the 32x32 fp32 A-panel is staged once
// per block into LDS via async copies (double buffered).
// MODE 0: fp32 row-major out. MODE 1: bf16 out scattered to [B,H,L,d_k].
template <int MODE>
__global__ __launch_bounds__(256)
void gemm_xwt(const float* __restrict__ X, const float* __restrict__ W,
              const float* __restrict__ bias, float* __restrict__ outF,
              __bf16* __restrict__ outH, int M, int N, int K) {
  __shared__ float Xs[2][32][36];  // +4 pad: rows shift 4 banks

  const int lane = threadIdx.x & 31;
  const int wave = threadIdx.x >> 5;
  const int gt   = blockIdx.x * 8 + wave;
  const int ntn  = N >> 5;
  const int m0   = (gt / ntn) << 5;   // identical for all 8 waves of a block
  const int n0   = (gt % ntn) << 5;

  // async-copy mapping: thread t moves 16B = X[m0 + t/8][k0 + (t%8)*4 ..+3]
  const int prow = threadIdx.x >> 3;
  const int pcol = (threadIdx.x & 7) << 2;

  v8f acc[2][2];
#pragma unroll
  for (int j = 0; j < 2; ++j) {
    float bj = bias[n0 + j * 16 + (lane & 15)];
#pragma unroll
    for (int i = 0; i < 2; ++i)
#pragma unroll
      for (int r = 0; r < 8; ++r) acc[i][j][r] = bj;
  }

  const int nIter = K >> 5;
  async_cp_b128(lds_off(&Xs[0][prow][pcol]),
                X + (size_t)(m0 + prow) * K + pcol);

  for (int it = 0; it < nIter; ++it) {
    const int buf = it & 1;
    const int k0  = it << 5;
    if (it + 1 < nIter) {
      async_cp_b128(lds_off(&Xs[buf ^ 1][prow][pcol]),
                    X + (size_t)(m0 + prow) * K + (k0 + 32) + pcol);
      __builtin_prefetch(W + (size_t)(n0 + (lane & 15)) * K + k0 + 64, 0, 0);
    }
    // per-wave-unique B panel straight to registers (overlaps async wait)
    v16bf b0 = b_frag_f32(W, K, n0 +      (lane & 15), k0, lane);
    v16bf b1 = b_frag_f32(W, K, n0 + 16 + (lane & 15), k0, lane);

    if (it + 1 < nIter) wait_async_le1(); else wait_async_le0();
    __syncthreads();  // panel `buf` visible to all waves

    // A fragments from LDS: lane half h -> K in [16h,16h+16)
    const int kh = (lane >> 4) << 4;
    v16bf a0, a1;
#pragma unroll
    for (int j = 0; j < 8; ++j) {
      a0[j]     = (__bf16)Xs[buf][lane & 15][kh + j];
      a0[j + 8] = (__bf16)Xs[buf][lane & 15][8 + kh + j];
      a1[j]     = (__bf16)Xs[buf][16 + (lane & 15)][kh + j];
      a1[j + 8] = (__bf16)Xs[buf][16 + (lane & 15)][8 + kh + j];
    }
    acc[0][0] = wmma_bf16(a0, b0, acc[0][0]);
    acc[0][1] = wmma_bf16(a0, b1, acc[0][1]);
    acc[1][0] = wmma_bf16(a1, b0, acc[1][0]);
    acc[1][1] = wmma_bf16(a1, b1, acc[1][1]);
    __syncthreads();  // everyone done with `buf` before it is re-filled
  }

#pragma unroll
  for (int i = 0; i < 2; ++i)
#pragma unroll
    for (int j = 0; j < 2; ++j)
#pragma unroll
      for (int r = 0; r < 8; ++r) {
        int m = m0 + i * 16 + r + ((lane >> 4) << 3);  // C/D: row r + 8*half
        int n = n0 + j * 16 + (lane & 15);
        float v = acc[i][j][r];
        if (MODE == 0) {
          outF[(size_t)m * N + n] = v;
        } else {
          int b = m >> 11, ql = m & 2047;   // m = b*2048 + q
          int h = n >> 6,  d  = n & 63;     // n = h*64 + d
          outH[(((size_t)b * NH + h) * L_SEQ + ql) * D_HEAD + d] = (__bf16)v;
        }
      }
}

// One block = one (b,h, 32-query stripe). S[32][2052] fp32 scores in LDS
// (pad 4 -> 16 consecutive rows land on distinct banks for phase-C reads).
// Vt: double-buffered 32x64 bf16 V panel staged by async copies.
__global__ __launch_bounds__(256)
void attn_topk(const __bf16* __restrict__ Qh, const __bf16* __restrict__ Kh,
               const __bf16* __restrict__ Vh, float* __restrict__ ctx) {
  __shared__ float  S[32][2052];      // 256.5 KB
  __shared__ __bf16 Vt[2][32 * 64];   //   8.0 KB   (total < 320 KB/WGP)

  const int lane = threadIdx.x & 31;
  const int wave = threadIdx.x >> 5;
  const int qt   = blockIdx.x & 63;
  const int bh   = blockIdx.x >> 6;  // b*NH + h
  const int q0   = qt * 32;
  const __bf16* Q  = Qh + (size_t)bh * L_SEQ * D_HEAD;
  const __bf16* Kp = Kh + (size_t)bh * L_SEQ * D_HEAD;
  const __bf16* Vp = Vh + (size_t)bh * L_SEQ * D_HEAD;

  // ---------- phase A: S = (Qtile @ K^T) / sqrt(d_k), 512 WMMAs/block ------
  v16bf aq[2][2];
#pragma unroll
  for (int qi = 0; qi < 2; ++qi)
#pragma unroll
    for (int c = 0; c < 2; ++c)
      aq[qi][c] = a_frag_bf16(Q, D_HEAD, q0 + qi * 16 + (lane & 15), c * 32, lane);

  for (int i = 0; i < 16; ++i) {
    int n0 = (wave + i * 8) << 4;  // key-tile base
    v16bf bk0 = b_frag_bf16(Kp, D_HEAD, n0 + (lane & 15), 0,  lane);
    v16bf bk1 = b_frag_bf16(Kp, D_HEAD, n0 + (lane & 15), 32, lane);
#pragma unroll
    for (int qi = 0; qi < 2; ++qi) {
      v8f acc;
#pragma unroll
      for (int r = 0; r < 8; ++r) acc[r] = 0.f;
      acc = wmma_bf16(aq[qi][0], bk0, acc);
      acc = wmma_bf16(aq[qi][1], bk1, acc);
#pragma unroll
      for (int r = 0; r < 8; ++r)
        S[qi * 16 + r + ((lane >> 4) << 3)][n0 + (lane & 15)] = acc[r] * 0.125f;
    }
  }
  __syncthreads();

  // ---------- phase B: per-row top-1024 threshold (bisection) + softmax ----
  for (int rr = 0; rr < 4; ++rr) {
    float* s = &S[wave * 4 + rr][0];
    float mx = -3.4e38f, mn = 3.4e38f;
    for (int i = lane; i < L_SEQ; i += 32) {
      float v = s[i]; mx = fmaxf(mx, v); mn = fminf(mn, v);
    }
#pragma unroll
    for (int o = 16; o; o >>= 1) {
      mx = fmaxf(mx, __shfl_xor(mx, o, 32));
      mn = fminf(mn, __shfl_xor(mn, o, 32));
    }
    float lo = mn, hi = mx;  // invariant: count(>=lo) >= TOPK > count(>=hi)
    for (int it = 0; it < 32; ++it) {
      float t = 0.5f * (lo + hi);
      int c = 0;
      for (int i = lane; i < L_SEQ; i += 32) c += (s[i] >= t);
#pragma unroll
      for (int o = 16; o; o >>= 1) c += __shfl_xor(c, o, 32);
      if (c >= TOPK) lo = t; else hi = t;
    }
    float thr = lo;
    float sum = 0.f;
    for (int i = lane; i < L_SEQ; i += 32) {
      float v = s[i];
      if (v >= thr) sum += __expf(v - mx);
    }
#pragma unroll
    for (int o = 16; o; o >>= 1) sum += __shfl_xor(sum, o, 32);
    float inv = 1.0f / sum;
    for (int i = lane; i < L_SEQ; i += 32) {
      float v = s[i];
      s[i] = (v >= thr) ? __expf(v - mx) * inv : 0.f;  // probs in place
    }
  }
  __syncthreads();

  // ---------- phase C: ctx[32,64] = P @ V, one 16x16 tile per wave ---------
  // 32-key x 64-d bf16 V panel (4KB) shared by all 8 waves: async-staged.
  const int qi = wave >> 2, ct = wave & 3;
  const int n0 = ct << 4;
  v8f acc;
#pragma unroll
  for (int r = 0; r < 8; ++r) acc[r] = 0.f;
  const int arow = qi * 16 + (lane & 15);
  const int dcol = n0 + (lane & 15);
  const unsigned vt0 = lds_off(&Vt[0][0]) + threadIdx.x * 16;  // 16B/thread
  const char*    vg  = (const char*)Vp + threadIdx.x * 16;

  async_cp_b128(vt0, vg);  // panel 0
  const int nIter = L_SEQ >> 5;
  for (int it = 0; it < nIter; ++it) {
    const int buf = it & 1;
    const int kk  = it << 5;
    if (it + 1 < nIter)
      async_cp_b128(vt0 + (buf ^ 1) * (32 * 64 * 2), vg + (size_t)(kk + 32) * 128);
    if (it + 1 < nIter) wait_async_le1(); else wait_async_le0();
    __syncthreads();  // panel `buf` visible

    int kb = kk + ((lane >> 4) << 4);
    v16bf a;
#pragma unroll
    for (int j = 0; j < 8; ++j) {
      a[j]     = (__bf16)S[arow][kb + j];
      a[j + 8] = (__bf16)S[arow][kb + 8 + j];
    }
    const __bf16* vp = &Vt[buf][((lane >> 4) << 4) * 64 + dcol];
    v16bf b;
#pragma unroll
    for (int e = 0; e < 16; ++e) b[e] = vp[e * 64];
    acc = wmma_bf16(a, b, acc);
    __syncthreads();  // all waves done with `buf` before refill
  }
  const int b_ = bh >> 4, h = bh & 15;
#pragma unroll
  for (int r = 0; r < 8; ++r) {
    int q = q0 + qi * 16 + r + ((lane >> 4) << 3);
    ctx[((size_t)(b_ * L_SEQ + q)) * D_MODEL + h * D_HEAD + n0 + (lane & 15)] = acc[r];
  }
}

extern "C" void kernel_launch(void* const* d_in, const int* in_sizes, int n_in,
                              void* d_out, int out_size, void* d_ws, size_t ws_size,
                              hipStream_t stream) {
  const float* q  = (const float*)d_in[0];
  const float* k  = (const float*)d_in[1];
  const float* v  = (const float*)d_in[2];
  const float* Wq = (const float*)d_in[3];
  const float* bq = (const float*)d_in[4];
  const float* Wk = (const float*)d_in[5];
  const float* bk = (const float*)d_in[6];
  const float* Wv = (const float*)d_in[7];
  const float* bv = (const float*)d_in[8];
  const float* Wo = (const float*)d_in[9];
  const float* bo = (const float*)d_in[10];

  const int M = 2 * L_SEQ, N = D_MODEL, K = D_MODEL;
  const size_t nh_elems = (size_t)2 * NH * L_SEQ * D_HEAD;  // 4M elems

  __bf16* Qh  = (__bf16*)d_ws;
  __bf16* Kh  = Qh + nh_elems;
  __bf16* Vh  = Kh + nh_elems;
  float*  ctx = (float*)(Vh + nh_elems);  // [B, L, D] fp32

  const int gemm_blocks = (M / 32) * (N / 32) / 8;  // 512

  gemm_xwt<1><<<gemm_blocks, 256, 0, stream>>>(q, Wq, bq, nullptr, Qh, M, N, K);
  gemm_xwt<1><<<gemm_blocks, 256, 0, stream>>>(k, Wk, bk, nullptr, Kh, M, N, K);
  gemm_xwt<1><<<gemm_blocks, 256, 0, stream>>>(v, Wv, bv, nullptr, Vh, M, N, K);

  attn_topk<<<2 * NH * (L_SEQ / 32), 256, 0, stream>>>(Qh, Kh, Vh, ctx);

  gemm_xwt<0><<<gemm_blocks, 256, 0, stream>>>(ctx, Wo, bo, (float*)d_out, nullptr, M, N, K);
}
